// HigherOrderGCNLayer_21466246545525
// MI455X (gfx1250) — matrix operand order repaired and checked by reading, not compiled, not run on hardware
//
#include <hip/hip_runtime.h>

#define NN     8192
#define FDIM   128
#define ORD    2
#define KSPLIT 4          // K-dimension split for occupancy (256 WGs in SpMM)

typedef __attribute__((ext_vector_type(16))) __bf16 v16bf;
typedef __attribute__((ext_vector_type(8)))  float  v8f;

struct U4 { unsigned x, y, z, w; };
struct F4 { float x, y, z, w; };

union Frag {
    unsigned u[8];
    U4       q[2];
    v16bf    v;
};

// round-to-nearest-even f32 -> bf16
static __device__ __forceinline__ unsigned short f2bf(float f) {
    union { float f; unsigned u; } c; c.f = f;
    unsigned r = c.u + 0x7FFFu + ((c.u >> 16) & 1u);
    return (unsigned short)(r >> 16);
}

// ---------------------------------------------------------------------------
// K0a: x (f32) -> bf16
// ---------------------------------------------------------------------------
__global__ __launch_bounds__(256) void hog_cvt_x(const float* __restrict__ x,
                                                 unsigned short* __restrict__ xbf) {
    int i = blockIdx.x * 256 + threadIdx.x;
    if (i < NN * FDIM) xbf[i] = f2bf(x[i]);
}

// ---------------------------------------------------------------------------
// K0b: W (f32, [K][N]) -> Wt bf16 [N][K]
// ---------------------------------------------------------------------------
__global__ __launch_bounds__(256) void hog_cvt_w(const float* __restrict__ W,
                                                 unsigned short* __restrict__ wt) {
    int i = blockIdx.x * 256 + threadIdx.x;
    if (i < FDIM * FDIM) {
        int n = i / FDIM, k = i % FDIM;
        wt[i] = f2bf(W[k * FDIM + n]);
    }
}

// ---------------------------------------------------------------------------
// K2: stream adjacency (512 MB) once -> bitmask (1 bit/entry, diag forced 1)
//     + degree -> dinv = rsqrt(deg).  One WG of 8 waves per (order,row).
// ---------------------------------------------------------------------------
__global__ __launch_bounds__(256) void hog_degbits(const float* __restrict__ adj,
                                                   unsigned* __restrict__ bits,
                                                   float* __restrict__ dinv) {
    __shared__ int scnt[8];
    int row = blockIdx.x;
    int o   = blockIdx.y;
    int tid = threadIdx.x;
    int w = tid >> 5, lane = tid & 31;

    const float* arow = adj + ((size_t)o * NN + row) * NN;
    unsigned*    brow = bits + (((size_t)o * NN + row) << 8);  // 256 dwords/row

    int cnt = 0;
    #pragma unroll 4
    for (int it = 0; it < 32; ++it) {
        int kc  = w * 32 + it;          // dword index within row
        int col = kc * 32 + lane;
        float v = arow[col];
        bool pred = (v != 0.0f) || (col == row);   // binarize + self loop
        unsigned msk = __builtin_amdgcn_ballot_w32(pred);
        if (lane == 0) brow[kc] = msk;
        cnt += __popc(msk);
    }
    if (lane == 0) scnt[w] = cnt;
    __syncthreads();
    if (tid == 0) {
        int deg = 0;
        #pragma unroll
        for (int i = 0; i < 8; ++i) deg += scnt[i];
        dinv[o * NN + row] = rsqrtf((float)deg);
    }
}

// ---------------------------------------------------------------------------
// K3: xw = x @ W via WMMA bf16; epilogue writes yT[o][f][j] = dinv_o[j]*xw[j][f]
//     (transposed bf16 so the SpMM B loads are K-contiguous b128 loads).
// ---------------------------------------------------------------------------
__global__ __launch_bounds__(256) void hog_ytbuild(const unsigned short* __restrict__ xbf,
                                                   const unsigned short* __restrict__ wt,
                                                   const float* __restrict__ dinv,
                                                   unsigned short* __restrict__ yT) {
    int row_base = blockIdx.x * 16;
    int tid  = threadIdx.x;
    int w    = tid >> 5;
    int lane = tid & 31;
    int h    = (lane >> 4) & 1;
    int m    = lane & 15;
    int col  = w * 16 + m;              // output column (n) for B and D

    v8f acc = {};
    #pragma unroll
    for (int kc = 0; kc < FDIM / 32; ++kc) {
        int k0 = kc * 32;
        Frag A, B;
        // A layout 16x32 bf16: lane(m,h); v0..3: K=8h+2v,+1 ; v4..7: K=16+8h+2(v-4),+1
        const unsigned short* ab = xbf + (size_t)(row_base + m) * FDIM + k0;
        A.q[0] = *(const U4*)(ab + h * 8);
        A.q[1] = *(const U4*)(ab + 16 + h * 8);
        // B layout 32x16 bf16: lane(n=m,h); v0..3: K=8h+2v,+1 ; v4..7: K=16+8h+2(v-4),+1
        const unsigned short* bb = wt + (size_t)col * FDIM + k0;
        B.q[0] = *(const U4*)(bb + h * 8);
        B.q[1] = *(const U4*)(bb + 16 + h * 8);
        acc = __builtin_amdgcn_wmma_f32_16x16x32_bf16(false, A.v, false, B.v,
                                                      (short)0, acc, false, false);
    }

    // D layout: lane n=m, VGPR r -> row m_out = r + 8h  (j = row_base + 8h + r)
    #pragma unroll
    for (int o = 0; o < ORD; ++o) {
        const F4* dvp = (const F4*)(dinv + o * NN + row_base + h * 8);
        F4 d0 = dvp[0], d1 = dvp[1];
        float dv[8] = { d0.x, d0.y, d0.z, d0.w, d1.x, d1.y, d1.z, d1.w };
        unsigned short q[8];
        #pragma unroll
        for (int r = 0; r < 8; ++r) q[r] = f2bf(dv[r] * acc[r]);
        U4 st;
        st.x = (unsigned)q[0] | ((unsigned)q[1] << 16);
        st.y = (unsigned)q[2] | ((unsigned)q[3] << 16);
        st.z = (unsigned)q[4] | ((unsigned)q[5] << 16);
        st.w = (unsigned)q[6] | ((unsigned)q[7] << 16);
        *(U4*)(yT + (size_t)(o * FDIM + col) * NN + row_base + h * 8) = st;
    }
}

// ---------------------------------------------------------------------------
// K4: bitmask-driven WMMA SpMM, partial over a K-slice.
//     Grid (64 M-blocks, KSPLIT).  Wave w owns rows [128*bx + 16w, +16) and ALL
//     128 output columns (8 N-tiles): ONE bit-expansion feeds 8 WMMAs, and the
//     bf16 B-panel is read from L2 only once per (M-block, K-slice).
//     part[s][i,f] = sum_o alpha_o * dinv_o[i] * (A_o @ yT_o)[i,f]  (K-slice s)
// ---------------------------------------------------------------------------
__global__ __launch_bounds__(256) void hog_spmm(const unsigned* __restrict__ bits,
                                                const float* __restrict__ dinv,
                                                const unsigned short* __restrict__ yT,
                                                const float* __restrict__ alpha,
                                                float* __restrict__ part) {
    int tid  = threadIdx.x;
    int w    = tid >> 5;
    int lane = tid & 31;
    int h    = (lane >> 4) & 1;
    int m    = lane & 15;
    int row_base = blockIdx.x * 128 + w * 16;
    int s        = blockIdx.y;                   // K-slice
    int kc_lo    = s * (NN / 32 / KSPLIT);       // 64 bitmask dwords per slice
    int kc_hi    = kc_lo + (NN / 32 / KSPLIT);

    float a0 = alpha[0], a1 = alpha[1];

    v8f fin[8];
    #pragma unroll
    for (int n = 0; n < 8; ++n) fin[n] = v8f{};

    #pragma unroll
    for (int o = 0; o < ORD; ++o) {
        const unsigned* brow = bits + (((size_t)o * NN + row_base + m) << 8);
        const unsigned short* bbase = yT + ((size_t)o * FDIM + m) * NN;

        v8f acc[8];
        #pragma unroll
        for (int n = 0; n < 8; ++n) acc[n] = v8f{};

        #pragma unroll 1
        for (int kc = kc_lo; kc < kc_hi; ++kc) {
            unsigned mask = brow[kc];
            // expand 16 bits -> bf16 {0,1} in the 16x32 A layout (shared by 8 WMMAs)
            Frag A;
            unsigned t = mask >> (h * 8);
            #pragma unroll
            for (int v = 0; v < 4; ++v) {
                unsigned lo = (t >> (2 * v)) & 1u;
                unsigned hi = (t >> (2 * v + 1)) & 1u;
                A.u[v] = (lo ? 0x3F80u : 0u) | (hi ? 0x3F800000u : 0u);
                lo = (t >> (16 + 2 * v)) & 1u;
                hi = (t >> (17 + 2 * v)) & 1u;
                A.u[v + 4] = (lo ? 0x3F80u : 0u) | (hi ? 0x3F800000u : 0u);
            }
            const unsigned short* bp = bbase + kc * 32 + h * 8;
            #pragma unroll
            for (int n = 0; n < 8; ++n) {
                Frag B;   // B tile n lives at constant offset n*16*NN -> imm offsets
                B.q[0] = *(const U4*)(bp + (size_t)n * 16 * NN);
                B.q[1] = *(const U4*)(bp + (size_t)n * 16 * NN + 16);
                acc[n] = __builtin_amdgcn_wmma_f32_16x16x32_bf16(false, A.v, false, B.v,
                                                                 (short)0, acc[n], false, false);
            }
        }

        float al = (o == 0) ? a0 : a1;
        const F4* dvp = (const F4*)(dinv + o * NN + row_base + h * 8);
        F4 d0 = dvp[0], d1 = dvp[1];
        float dv[8] = { d0.x, d0.y, d0.z, d0.w, d1.x, d1.y, d1.z, d1.w };
        #pragma unroll
        for (int n = 0; n < 8; ++n)
            #pragma unroll
            for (int r = 0; r < 8; ++r)
                fin[n][r] += al * dv[r] * acc[n][r];
    }

    float* pbase = part + (size_t)s * NN * FDIM;
    #pragma unroll
    for (int n = 0; n < 8; ++n) {
        int col = n * 16 + m;
        #pragma unroll
        for (int r = 0; r < 8; ++r) {
            int row = row_base + h * 8 + r;
            pbase[(size_t)row * FDIM + col] = fin[n][r];
        }
    }
}

// ---------------------------------------------------------------------------
// K5: deterministic reduction of K-slice partials + bias term
// ---------------------------------------------------------------------------
__global__ __launch_bounds__(256) void hog_reduce(const float* __restrict__ part,
                                                  const float* __restrict__ bvec,
                                                  const float* __restrict__ alpha,
                                                  float* __restrict__ out) {
    int i = blockIdx.x * 256 + threadIdx.x;
    if (i >= NN * FDIM) return;
    float acc = 0.0f;
    #pragma unroll
    for (int sIdx = 0; sIdx < KSPLIT; ++sIdx)
        acc += part[(size_t)sIdx * NN * FDIM + i];
    out[i] = acc + (alpha[0] + alpha[1]) * bvec[i & (FDIM - 1)];
}

// ---------------------------------------------------------------------------
// Launch
// ---------------------------------------------------------------------------
extern "C" void kernel_launch(void* const* d_in, const int* in_sizes, int n_in,
                              void* d_out, int out_size, void* d_ws, size_t ws_size,
                              hipStream_t stream) {
    const float* x     = (const float*)d_in[0];
    // d_in[1] edge_index: unused by the math
    const float* adj   = (const float*)d_in[2];
    const float* W     = (const float*)d_in[3];
    const float* bvec  = (const float*)d_in[4];
    const float* alpha = (const float*)d_in[5];
    float* out = (float*)d_out;

    char* ws = (char*)d_ws;
    const size_t OFF_BITS = 0;                                     // 16 MB
    const size_t OFF_DINV = OFF_BITS + (size_t)ORD * NN * 256 * 4; // 64 KB
    const size_t OFF_YT   = OFF_DINV + (size_t)ORD * NN * 4;       // 4 MB
    const size_t OFF_XBF  = OFF_YT   + (size_t)ORD * FDIM * NN * 2;// 2 MB
    const size_t OFF_WT   = OFF_XBF  + (size_t)NN * FDIM * 2;      // 32 KB
    const size_t OFF_PART = OFF_WT   + (size_t)FDIM * FDIM * 2;    // 16 MB

    unsigned*       bits = (unsigned*)(ws + OFF_BITS);
    float*          dinv = (float*)(ws + OFF_DINV);
    unsigned short* yT   = (unsigned short*)(ws + OFF_YT);
    unsigned short* xbf  = (unsigned short*)(ws + OFF_XBF);
    unsigned short* wt   = (unsigned short*)(ws + OFF_WT);
    float*          part = (float*)(ws + OFF_PART);

    hog_cvt_x<<<(NN * FDIM + 255) / 256, 256, 0, stream>>>(x, xbf);
    hog_cvt_w<<<(FDIM * FDIM + 255) / 256, 256, 0, stream>>>(W, wt);

    dim3 g2(NN, ORD);
    hog_degbits<<<g2, 256, 0, stream>>>(adj, bits, dinv);

    hog_ytbuild<<<NN / 16, 256, 0, stream>>>(xbf, wt, dinv, yT);

    dim3 g4(NN / 128, KSPLIT);
    hog_spmm<<<g4, 256, 0, stream>>>(bits, dinv, yT, alpha, part);

    hog_reduce<<<(NN * FDIM + 255) / 256, 256, 0, stream>>>(part, bvec, alpha, out);
}
